// PoetryModel_2697239462759
// MI455X (gfx1250) — compile-verified, matrix operand order checked by reading, not run on hardware
//
#include <hip/hip_runtime.h>

typedef __attribute__((ext_vector_type(16))) _Float16 v16h;
typedef __attribute__((ext_vector_type(8)))  _Float16 v8h;
typedef __attribute__((ext_vector_type(8)))  float    v8f;
typedef __attribute__((ext_vector_type(4)))  unsigned int u32x4;
typedef __attribute__((ext_vector_type(8)))  int          i32x8;
typedef __attribute__((ext_vector_type(4)))  int          i32x4;

#define HID    1024
#define EMB    512
#define SEQL   128
#define BATCH  64
#define NVOCAB 32000
#define G4H    4096   /* 4*HID */

#if __has_builtin(__builtin_amdgcn_tensor_load_to_lds)
#define HAVE_TDM 1
#else
#define HAVE_TDM 0
#endif

// ---------------------------------------------------------------------------
// Fragment loader for V_WMMA_F32_16X16X32_F16 (wave32), from a row-major
// (ld = row pitch in halfs) source. A: lane L holds row M=L&15;
//   halfs[0..7]  = [row, kb .. kb+7],  halfs[8..15] = [row, kb+16 .. kb+23],
//   kb = (lane>>4)*8.  B column-per-lane uses identical packing on Bt (N x K).
// ---------------------------------------------------------------------------
__device__ __forceinline__ v16h load_frag(const _Float16* __restrict__ base,
                                          int ld, int r0, int k0, int lane) {
  const int row = r0 + (lane & 15);
  const int kb  = k0 + ((lane >> 4) << 3);
  const _Float16* p = base + (size_t)row * (size_t)ld + kb;
  v8h lo = *(const v8h*)p;
  v8h hi = *(const v8h*)(p + 16);
  v16h r;
#pragma unroll
  for (int i = 0; i < 8; ++i) { r[i] = lo[i]; r[i + 8] = hi[i]; }
  return r;
}

// Same fragment gather but from an LDS tile with fixed pitch of 32 halfs.
__device__ __forceinline__ v16h lds_frag(const _Float16* buf, int r0, int lane) {
  const int row = r0 + (lane & 15);
  const int kb  = (lane >> 4) << 3;
  const _Float16* p = buf + row * 32 + kb;
  v8h lo = *(const v8h*)p;          // -> ds_load_b128
  v8h hi = *(const v8h*)(p + 16);
  v16h r;
#pragma unroll
  for (int i = 0; i < 8; ++i) { r[i] = lo[i]; r[i + 8] = hi[i]; }
  return r;
}

// ---------------------------------------------------------------------------
// Tensor Data Mover: DMA a (rows x 32) f16 tile (row pitch ld_elems) from
// global into LDS at byte offset lds_off.  D# layout per CDNA5 ISA ch.8:
//   group0: count=1 | lds_addr | global_addr | type=2
//   group1: data_size=1(2B), tensor_dim0=ld, tile_dim0=32, tile_dim1=rows,
//           tensor_dim0_stride=ld
// ---------------------------------------------------------------------------
__device__ __forceinline__ void tdm_load_tile_f16(unsigned lds_off,
                                                  const _Float16* gptr,
                                                  int rows, int ld_elems) {
#if HAVE_TDM
  unsigned long long ga = (unsigned long long)(size_t)gptr;
  u32x4 g0;
  g0[0] = 1u;                                            // count=1 (valid), user
  g0[1] = lds_off;                                       // LDS byte address
  g0[2] = (unsigned)(ga & 0xFFFFFFFFull);                // global_addr[31:0]
  g0[3] = (unsigned)((ga >> 32) & 0x1FFFFFFu) | (2u << 30); // [56:32] | type=2

  const unsigned td0 = (unsigned)ld_elems;               // tensor_dim0 (elems)
  const unsigned td1 = 0x7FFFFFFFu;                      // tensor_dim1: no clip
  const unsigned long long s0 = (unsigned long long)ld_elems; // dim0 stride
  i32x8 g1;
  g1[0] = (int)(1u << 16);                               // mask=0, data_size=1
  g1[1] = (int)((td0 & 0xFFFFu) << 16);                  // dim0[15:0]
  g1[2] = (int)((td0 >> 16) | ((td1 & 0xFFFFu) << 16));  // dim0[31:16]|dim1[15:0]
  g1[3] = (int)((td1 >> 16) | (32u << 16));              // dim1[31:16]|tile_dim0
  g1[4] = (int)((unsigned)rows);                         // tile_dim1 | tile_dim2=0
  g1[5] = (int)(unsigned)(s0 & 0xFFFFFFFFull);           // stride0[31:0]
  g1[6] = (int)(unsigned)((s0 >> 32) & 0xFFFFu);         // stride0[47:32]|s1=0
  g1[7] = 0;
  i32x4 z4 = {0, 0, 0, 0};
#if defined(__clang_major__) && (__clang_major__ >= 23)
  i32x8 z8 = {0, 0, 0, 0, 0, 0, 0, 0};
  __builtin_amdgcn_tensor_load_to_lds(g0, g1, z4, z4, z8, 0);
#else
  __builtin_amdgcn_tensor_load_to_lds(g0, g1, z4, z4, 0);
#endif
#endif
}

#if !HAVE_TDM
// Cooperative fallback: whole block copies a (rows x 32) f16 tile into LDS.
__device__ __forceinline__ void coop_copy_tile(_Float16* dst, const _Float16* src,
                                               int rows, int ld_elems,
                                               int tid, int nthreads) {
  const int chunks = rows * 4;                 // v8h chunks per tile
  for (int ci = tid; ci < chunks; ci += nthreads) {
    int r = ci >> 2, c = (ci & 3) << 3;
    *(v8h*)&dst[r * 32 + c] = *(const v8h*)&src[(size_t)r * ld_elems + c];
  }
}
#endif

// ---------------------------------------------------------------------------
// TDM/LDS-tiled WMMA GEMM:
// C[M,N] = A[M,K](f16 rm) * Bt[N,K](f16 rm) (+bias[n]) (+addend[m,n])
// Block = 256 thr = 8 waves (2Mx4N), block tile 64(M) x 256(N), K-chunk 32.
// Wave0 issues TDM DMAs for both tiles, waits TENSORcnt, barrier publishes.
// ---------------------------------------------------------------------------
__global__ void __launch_bounds__(256)
wmma_gemm_f16_lds(const _Float16* __restrict__ A, const _Float16* __restrict__ Bt,
                  float* __restrict__ C, const float* __restrict__ bias,
                  const float* __restrict__ addend, int M, int N, int K)
{
  __shared__ _Float16 Asm[64 * 32];     // 4 KB
  __shared__ _Float16 Bsm[256 * 32];    // 16 KB
  const int lane = threadIdx.x & 31;
  const int wid  = threadIdx.x >> 5;
  const int mblk = blockIdx.y * 64;
  const int nblk = blockIdx.x * 256;
  const int m0 = (wid >> 2) * 32;       // wave tile origin (local)
  const int n0 = (wid & 3) * 64;

  // Low 32 bits of the generic address of an LDS object == LDS byte offset
  // (flat aperture: the LDS address is the low 32 bits, ISA 10.2).
  const unsigned aoff = (unsigned)(size_t)(void*)Asm;
  const unsigned boff = (unsigned)(size_t)(void*)Bsm;

  v8f acc[2][4] = {};

  for (int k0 = 0; k0 < K; k0 += 32) {
    __syncthreads();                    // previous chunk's reads complete
#if HAVE_TDM
    if (wid == 0) {                     // wave-uniform issue; EXEC ignored by TDM
      tdm_load_tile_f16(aoff, A  + (size_t)mblk * K + k0,  64, K);
      tdm_load_tile_f16(boff, Bt + (size_t)nblk * K + k0, 256, K);
      __builtin_amdgcn_s_wait_tensorcnt(0);
    }
#else
    coop_copy_tile(Asm, A  + (size_t)mblk * K + k0,  64, K, threadIdx.x, 256);
    coop_copy_tile(Bsm, Bt + (size_t)nblk * K + k0, 256, K, threadIdx.x, 256);
#endif
    __syncthreads();                    // tiles visible to all 8 waves

    v16h a0 = lds_frag(Asm, m0, lane);
    v16h a1 = lds_frag(Asm, m0 + 16, lane);
#pragma unroll
    for (int j = 0; j < 4; ++j) {
      v16h b = lds_frag(Bsm, n0 + 16 * j, lane);
      acc[0][j] = __builtin_amdgcn_wmma_f32_16x16x32_f16(
          false, a0, false, b, (short)0, acc[0][j], false, false);
      acc[1][j] = __builtin_amdgcn_wmma_f32_16x16x32_f16(
          false, a1, false, b, (short)0, acc[1][j], false, false);
    }
  }

  const int mlo = (lane >> 4) << 3;
  const int nn  = lane & 15;
#pragma unroll
  for (int i = 0; i < 2; ++i) {
    const int mb = mblk + m0 + i * 16 + mlo;
#pragma unroll
    for (int j = 0; j < 4; ++j) {
      const int n = nblk + n0 + j * 16 + nn;
      const float bn = bias ? bias[n] : 0.0f;
#pragma unroll
      for (int r = 0; r < 8; ++r) {
        const size_t off = (size_t)(mb + r) * (size_t)N + n;
        float v = acc[i][j][r] + bn;
        if (addend) v += addend[off];
        C[off] = v;
      }
    }
  }
}

// ---------------------------------------------------------------------------
// Direct-from-global WMMA GEMM (used for the 128 serial recurrence steps:
// W_hh^T is L2-resident, and avoiding barriers shortens the serial chain).
// ---------------------------------------------------------------------------
__global__ void __launch_bounds__(256)
wmma_gemm_f16(const _Float16* __restrict__ A, const _Float16* __restrict__ Bt,
              float* __restrict__ C, const float* __restrict__ bias,
              const float* __restrict__ addend, int M, int N, int K)
{
  const int lane = threadIdx.x & 31;
  const int wid  = threadIdx.x >> 5;
  const int m0 = blockIdx.y * 64  + (wid >> 2) * 32;
  const int n0 = blockIdx.x * 256 + (wid & 3) * 64;
  if (m0 >= M || n0 >= N) return;     // uniform: EXEC stays all-1s for WMMA

  v8f acc[2][4] = {};
  for (int k0 = 0; k0 < K; k0 += 32) {
    if (k0 + 32 < K)
      __builtin_prefetch(A + (size_t)(m0 + (lane & 15)) * K + k0 + 32, 0, 0);
    v16h a0 = load_frag(A, K, m0,      k0, lane);
    v16h a1 = load_frag(A, K, m0 + 16, k0, lane);
#pragma unroll
    for (int j = 0; j < 4; ++j) {
      v16h b = load_frag(Bt, K, n0 + 16 * j, k0, lane);
      acc[0][j] = __builtin_amdgcn_wmma_f32_16x16x32_f16(
          false, a0, false, b, (short)0, acc[0][j], false, false);
      acc[1][j] = __builtin_amdgcn_wmma_f32_16x16x32_f16(
          false, a1, false, b, (short)0, acc[1][j], false, false);
    }
  }
  const int mlo = (lane >> 4) << 3;
  const int nn  = lane & 15;
#pragma unroll
  for (int i = 0; i < 2; ++i) {
    const int mb = m0 + i * 16 + mlo;
#pragma unroll
    for (int j = 0; j < 4; ++j) {
      const int n = n0 + j * 16 + nn;
      const float bn = bias ? bias[n] : 0.0f;
#pragma unroll
      for (int r = 0; r < 8; ++r) {
        const size_t off = (size_t)(mb + r) * (size_t)N + n;
        float v = acc[i][j][r] + bn;
        if (addend) v += addend[off];
        C[off] = v;
      }
    }
  }
}

// ---------------------------------------------------------------------------
// Helper kernels
// ---------------------------------------------------------------------------
__global__ void f32_to_f16_kernel(const float* __restrict__ in,
                                  _Float16* __restrict__ out, int n) {
  int i = blockIdx.x * blockDim.x + threadIdx.x;
  if (i < n) out[i] = (_Float16)in[i];
}

__global__ void transpose_f32_to_f16_kernel(const float* __restrict__ in,
                                            _Float16* __restrict__ out,
                                            int K, int N) {
  int i = blockIdx.x * blockDim.x + threadIdx.x;
  if (i < K * N) {
    int k = i / N, n = i - k * N;
    out[(size_t)n * K + k] = (_Float16)in[i];
  }
}

__global__ void embed_f16_kernel(const int* __restrict__ tok,
                                 const float* __restrict__ emb,
                                 _Float16* __restrict__ out, int n) {
  int i = blockIdx.x * blockDim.x + threadIdx.x;
  if (i < n) {
    int p = i >> 9;            // / EMB
    int e = i & (EMB - 1);
    out[i] = (_Float16)emb[(size_t)tok[p] * EMB + e];
  }
}

__device__ __forceinline__ float sigmoidf_(float x) {
  return 1.0f / (1.0f + __expf(-x));
}

__global__ void lstm_cell_kernel(const float* __restrict__ gates,
                                 float* __restrict__ h, float* __restrict__ c,
                                 _Float16* __restrict__ h_half,
                                 _Float16* __restrict__ hseq_t) {
  int idx = blockIdx.x * blockDim.x + threadIdx.x;   // BATCH*HID = 65536
  if (idx >= BATCH * HID) return;
  int b = idx >> 10;
  int j = idx & (HID - 1);
  const float* g = gates + (size_t)b * G4H;
  float ig = sigmoidf_(g[j]);
  float fg = sigmoidf_(g[HID + j]);
  float gg = tanhf(g[2 * HID + j]);
  float og = sigmoidf_(g[3 * HID + j]);
  float cn = fg * c[idx] + ig * gg;
  float hn = og * tanhf(cn);
  c[idx] = cn;
  h[idx] = hn;
  h_half[idx] = (_Float16)hn;
  hseq_t[idx] = (_Float16)hn;
}

// ---------------------------------------------------------------------------
// Launch
// ---------------------------------------------------------------------------
extern "C" void kernel_launch(void* const* d_in, const int* in_sizes, int n_in,
                              void* d_out, int out_size, void* d_ws, size_t ws_size,
                              hipStream_t stream) {
  const int*   inp    = (const int*)  d_in[0];  // (SEQ, BATCH)
  const float* emb    = (const float*)d_in[1];  // (VOCAB, EMB)
  const float* W_ih   = (const float*)d_in[2];  // (EMB, 4H)
  const float* W_hh   = (const float*)d_in[3];  // (HID, 4H)
  const float* b_lstm = (const float*)d_in[4];  // (4H)
  const float* lin_W  = (const float*)d_in[5];  // (VOCAB, HID) == Bt layout
  const float* lin_b  = (const float*)d_in[6];  // (VOCAB)
  const float* h0     = (const float*)d_in[7];  // (BATCH, HID)
  const float* c0     = (const float*)d_in[8];  // (BATCH, HID)
  float* out = (float*)d_out;                   // logits | h_t | c_t

  char* ws = (char*)d_ws;
  size_t off = 0;
  _Float16* xh    = (_Float16*)(ws + off); off += (size_t)SEQL * BATCH * EMB * 2;
  _Float16* WihT  = (_Float16*)(ws + off); off += (size_t)G4H * EMB * 2;
  _Float16* WhhT  = (_Float16*)(ws + off); off += (size_t)G4H * HID * 2;
  _Float16* linWh = (_Float16*)(ws + off); off += (size_t)NVOCAB * HID * 2;   // 64 MB, L2-resident
  float*    xg    = (float*)   (ws + off); off += (size_t)SEQL * BATCH * G4H * 4;
  float*    gates = (float*)   (ws + off); off += (size_t)BATCH * G4H * 4;
  _Float16* hseq  = (_Float16*)(ws + off); off += (size_t)SEQL * BATCH * HID * 2;
  float*    hS    = (float*)   (ws + off); off += (size_t)BATCH * HID * 4;
  float*    cS    = (float*)   (ws + off); off += (size_t)BATCH * HID * 4;
  _Float16* hHalf = (_Float16*)(ws + off); off += (size_t)BATCH * HID * 2;
  (void)ws_size; (void)in_sizes; (void)n_in; (void)out_size;

  const int T256 = 256;
  transpose_f32_to_f16_kernel<<<(EMB * G4H + T256 - 1) / T256, T256, 0, stream>>>(W_ih, WihT, EMB, G4H);
  transpose_f32_to_f16_kernel<<<(HID * G4H + T256 - 1) / T256, T256, 0, stream>>>(W_hh, WhhT, HID, G4H);
  f32_to_f16_kernel<<<((int)((size_t)NVOCAB * HID) + T256 - 1) / T256, T256, 0, stream>>>(lin_W, linWh, NVOCAB * HID);

  embed_f16_kernel<<<(SEQL * BATCH * EMB + T256 - 1) / T256, T256, 0, stream>>>(
      inp, emb, xh, SEQL * BATCH * EMB);

  // xg = x @ W_ih + b_lstm   (TDM/LDS-tiled path)
  {
    dim3 grid(G4H / 256, (SEQL * BATCH) / 64);
    wmma_gemm_f16_lds<<<grid, 256, 0, stream>>>(xh, WihT, xg, b_lstm, nullptr,
                                                SEQL * BATCH, G4H, EMB);
  }

  hipMemcpyAsync(hS, h0, (size_t)BATCH * HID * 4, hipMemcpyDeviceToDevice, stream);
  hipMemcpyAsync(cS, c0, (size_t)BATCH * HID * 4, hipMemcpyDeviceToDevice, stream);
  f32_to_f16_kernel<<<(BATCH * HID + T256 - 1) / T256, T256, 0, stream>>>(h0, hHalf, BATCH * HID);

  // 128 serial LSTM steps (direct WMMA path, W_hh^T streamed from L2)
  for (int t = 0; t < SEQL; ++t) {
    dim3 grid(G4H / 256, 1);
    wmma_gemm_f16<<<grid, 256, 0, stream>>>(hHalf, WhhT, gates, nullptr,
                                            xg + (size_t)t * BATCH * G4H,
                                            BATCH, G4H, HID);
    lstm_cell_kernel<<<(BATCH * HID + T256 - 1) / T256, T256, 0, stream>>>(
        gates, hS, cS, hHalf, hseq + (size_t)t * BATCH * HID);
  }

  // logits = hseq @ lin_W^T + lin_b  (TDM/LDS-tiled path, store-bound on HBM)
  {
    dim3 grid(NVOCAB / 256, (SEQL * BATCH) / 64);
    wmma_gemm_f16_lds<<<grid, 256, 0, stream>>>(hseq, linWh, out, lin_b, nullptr,
                                                SEQL * BATCH, NVOCAB, HID);
  }

  float* out_h = out + (size_t)SEQL * BATCH * NVOCAB;
  float* out_c = out_h + (size_t)BATCH * HID;
  hipMemcpyAsync(out_h, hS, (size_t)BATCH * HID * 4, hipMemcpyDeviceToDevice, stream);
  hipMemcpyAsync(out_c, cS, (size_t)BATCH * HID * 4, hipMemcpyDeviceToDevice, stream);
}
